// FeatureNet_57174604644986
// MI455X (gfx1250) — matrix operand order repaired and checked by reading, not compiled, not run on hardware
//
#include <hip/hip_runtime.h>

#define NB 2048   // batch
#define TT 50     // timesteps
#define NC 40     // codes per visit
#define ED 256    // embedding / LSTM units
#define G4 1024   // 4*ED
#define LDA 264   // padded LDS row stride in halfwords (264*2B = 33 x 16B, bank-spreading)

typedef __attribute__((ext_vector_type(16))) __bf16         v16bf;
typedef __attribute__((ext_vector_type(8)))  float          v8f;
typedef __attribute__((ext_vector_type(8)))  unsigned short ushort8;

union BfFrag { v16bf v; ushort8 h[2]; };

__device__ __forceinline__ unsigned short f2bf(float f) {
  union { float f; unsigned int u; } cv; cv.f = f;
  unsigned int u = cv.u;
  return (unsigned short)((u + 0x7FFFu + ((u >> 16) & 1u)) >> 16);  // RNE
}
__device__ __forceinline__ float bf2f(unsigned short u) {
  union { unsigned int u; float f; } cv; cv.u = ((unsigned int)u) << 16; return cv.f;
}
__device__ __forceinline__ float sigm(float x) { return 1.0f / (1.0f + __expf(-x)); }

// ---------------- weight prep: f32 -> bf16 (K-major for B operands) ----------------
__global__ void prep_weights(const float* __restrict__ Kf, const float* __restrict__ Rf,
                             const float* __restrict__ W0f, const float* __restrict__ W1f,
                             const float* __restrict__ Wlin,
                             unsigned short* __restrict__ Kt, unsigned short* __restrict__ Rt,
                             unsigned short* __restrict__ W0t, unsigned short* __restrict__ W1t,
                             unsigned short* __restrict__ Wb) {
  int idx = blockIdx.x * 256 + threadIdx.x;     // covers 256*1024
  int k = idx >> 10, n = idx & 1023;            // [256,1024] row-major source
  Kt[(size_t)n * ED + k] = f2bf(Kf[idx]);
  Rt[(size_t)n * ED + k] = f2bf(Rf[idx]);
  if (idx < ED * ED) {
    int kk = idx >> 8, nn = idx & 255;
    W0t[(size_t)nn * ED + kk] = f2bf(W0f[idx]);
    W1t[(size_t)nn * ED + kk] = f2bf(W1f[idx]);
  }
  if (idx < 536 * ED) Wb[idx] = f2bf(Wlin[idx]);   // row-major copy, halves embed traffic
}

__global__ void zero_state(float* __restrict__ c, unsigned short* __restrict__ h0) {
  int idx = blockIdx.x * 256 + threadIdx.x;
  c[idx] = 0.f; h0[idx] = 0;
}

// ---------------- sparse multi-hot embed + input linear + relu ----------------
__global__ void embed_kernel(const int* __restrict__ code, const float* __restrict__ aux,
                             const unsigned short* __restrict__ Wb,
                             const float* __restrict__ blin,
                             unsigned short* __restrict__ x_bf) {
  const int nt = blockIdx.x;       // (n*TT + t)
  const int tid = threadIdx.x;     // output channel
  __shared__ int sc[NC];
  __shared__ float sax[10];
  __shared__ unsigned char su[NC];
  if (tid < NC) sc[tid] = code[(size_t)nt * NC + tid];
  if (tid >= 64 && tid < 74) sax[tid - 64] = aux[(size_t)nt * 10 + (tid - 64)];
  __syncthreads();
  if (tid < NC) {  // binary CategoryEncoding: dedup repeated codes, drop code 0
    int c = sc[tid];
    unsigned char u = (c != 0);
    for (int j = 0; j < tid; ++j) if (sc[j] == c) u = 0;
    su[tid] = u;
  }
  __syncthreads();
  float acc = blin[tid];
  for (int j = 0; j < NC; ++j)
    if (su[j]) acc += bf2f(Wb[(size_t)(sc[j] - 1) * ED + tid]);  // mh drops col 0
#pragma unroll
  for (int a = 0; a < 10; ++a)
    acc += sax[a] * bf2f(Wb[(size_t)(526 + a) * ED + tid]);
  x_bf[(size_t)nt * ED + tid] = f2bf(fmaxf(acc, 0.f));
}

// one k-chunk: load B frag + 4 A frags into distinct registers, then 4 WMMAs.
// Distinct frags let the in-order LDS returns be consumed with partial dscnt waits.
__device__ __forceinline__ void gemm_chunk(const unsigned short* __restrict__ bcol,
                                           const unsigned short* __restrict__ sBase,
                                           int k0, int half, int l16, v8f acc[4]) {
  BfFrag b, a0, a1, a2, a3;
  b.h[0] = *(const ushort8*)(bcol + k0 + half * 8);
  b.h[1] = *(const ushort8*)(bcol + k0 + 16 + half * 8);
  const unsigned short* r0 = sBase + (0 * 16 + l16) * LDA + k0 + half * 8;
  const unsigned short* r1 = sBase + (1 * 16 + l16) * LDA + k0 + half * 8;
  const unsigned short* r2 = sBase + (2 * 16 + l16) * LDA + k0 + half * 8;
  const unsigned short* r3 = sBase + (3 * 16 + l16) * LDA + k0 + half * 8;
  a0.h[0] = *(const ushort8*)(r0); a0.h[1] = *(const ushort8*)(r0 + 16);
  a1.h[0] = *(const ushort8*)(r1); a1.h[1] = *(const ushort8*)(r1 + 16);
  a2.h[0] = *(const ushort8*)(r2); a2.h[1] = *(const ushort8*)(r2 + 16);
  a3.h[0] = *(const ushort8*)(r3); a3.h[1] = *(const ushort8*)(r3 + 16);
  acc[0] = __builtin_amdgcn_wmma_f32_16x16x32_bf16(false, a0.v, false, b.v, (short)0, acc[0], false, false);
  acc[1] = __builtin_amdgcn_wmma_f32_16x16x32_bf16(false, a1.v, false, b.v, (short)0, acc[1], false, false);
  acc[2] = __builtin_amdgcn_wmma_f32_16x16x32_bf16(false, a2.v, false, b.v, (short)0, acc[2], false, false);
  acc[3] = __builtin_amdgcn_wmma_f32_16x16x32_bf16(false, a3.v, false, b.v, (short)0, acc[3], false, false);
}

// ---------------- fused LSTM step ----------------
// Block tile: 64 batch rows x 32 e-cols x all 4 gates. A tiles (x_t, h) staged in LDS
// and shared by all waves; each wave owns a distinct (gate, e-subtile) B column strip
// and computes a 64x16 strip (4 WMMAs per B fragment). Gate pre-activations exchanged
// through LDS; z never touches HBM.
__global__ void lstm_step_fused(const unsigned short* __restrict__ x_bf, int t,
                                const unsigned short* __restrict__ h_in,
                                unsigned short* __restrict__ h_out,
                                float* __restrict__ c_state,
                                const unsigned short* __restrict__ Kt,
                                const unsigned short* __restrict__ Rt,
                                const float* __restrict__ b_lstm,
                                const int* __restrict__ length,
                                unsigned short* __restrict__ xlast_bf) {
  __shared__ unsigned short sA[64 * LDA];   // x_t tile (rows mBase..mBase+63)
  __shared__ unsigned short sH[64 * LDA];   // h_{t-1} tile
  __shared__ float zs[4][64][33];           // [gate][m][e], padded
  const int tid  = threadIdx.x;
  const int wave = tid >> 5;
  const int lane = tid & 31;
  const int half = lane >> 4;
  const int l16  = lane & 15;
  const int g    = wave & 3;                 // gate i/f/g/o
  const int esub = wave >> 2;                // e-subtile 0..1
  const int mBase = blockIdx.y * 64;
  const int col = g * 256 + blockIdx.x * 32 + esub * 16 + l16;   // col in 4E space

  // cooperative stage of A tiles into LDS (each b128 chunk loaded exactly once)
  {
    const unsigned short* xb = x_bf + (size_t)t * ED;   // row n lives at n*TT*ED
#pragma unroll
    for (int c = tid; c < 64 * 32; c += 256) {          // 64 rows x 32 chunks of 8 hw
      int row = c >> 5, kb = (c & 31) << 3;
      *(ushort8*)&sA[row * LDA + kb] =
          *(const ushort8*)(xb + (size_t)(mBase + row) * TT * ED + kb);
      *(ushort8*)&sH[row * LDA + kb] =
          *(const ushort8*)(h_in + (size_t)(mBase + row) * ED + kb);
    }
  }
  const unsigned short* bcol_K = Kt + (size_t)col * ED;
  const unsigned short* bcol_R = Rt + (size_t)col * ED;
  __builtin_prefetch(bcol_K, 0, 3);
  __builtin_prefetch(bcol_R, 0, 3);
  __syncthreads();

  v8f acc[4] = {};
#pragma unroll
  for (int k0 = 0; k0 < 256; k0 += 32)      // x_t @ K_lstm
    gemm_chunk(bcol_K, sA, k0, half, l16, acc);
#pragma unroll
  for (int k0 = 0; k0 < 256; k0 += 32)      // h_{t-1} @ R_lstm
    gemm_chunk(bcol_R, sH, k0, half, l16, acc);

  const float bias = b_lstm[col];
#pragma unroll
  for (int ms = 0; ms < 4; ++ms)
#pragma unroll
    for (int r = 0; r < 8; ++r)   // C layout: M = half*8 + r, N = l16
      zs[g][ms * 16 + half * 8 + r][esub * 16 + l16] = acc[ms][r] + bias;
  __syncthreads();

  // fused gate math: 64x32 = 2048 elements, 8 per thread
#pragma unroll
  for (int p = 0; p < 8; ++p) {
    int elem = tid + p * 256;
    int ml = elem >> 5, e = elem & 31;
    float zi = zs[0][ml][e];
    float zf = zs[1][ml][e];
    float zg = zs[2][ml][e];
    float zo = zs[3][ml][e];
    int n  = mBase + ml;
    int ge = blockIdx.x * 32 + e;
    size_t idx = (size_t)n * ED + ge;
    float cp = c_state[idx];
    float cn = sigm(zf) * cp + sigm(zi) * tanhf(zg);
    float hn = sigm(zo) * tanhf(cn);
    c_state[idx] = cn;
    unsigned short hb = f2bf(hn);
    h_out[idx] = hb;
    if (length[n] - 1 == t) xlast_bf[idx] = hb;  // fused gather of last valid step
  }
}

// ---------------- head GEMMs: 2048x256x256, bias (+relu), f32 and/or bf16 out ----------------
__global__ void head_gemm(const unsigned short* __restrict__ A,
                          const unsigned short* __restrict__ Bt,
                          const float* __restrict__ bias,
                          float* __restrict__ outF,
                          unsigned short* __restrict__ outB, int doRelu) {
  const int tid = threadIdx.x, wave = tid >> 5, lane = tid & 31;
  const int half = lane >> 4, l16 = lane & 15;
  const int m0 = blockIdx.y * 32 + (wave >> 2) * 16;
  const int n0 = blockIdx.x * 64 + (wave & 3) * 16;
  const unsigned short* arow = A + (size_t)(m0 + l16) * ED;
  const unsigned short* bcol = Bt + (size_t)(n0 + l16) * ED;
  v8f acc = {};
#pragma unroll
  for (int k0 = 0; k0 < 256; k0 += 32) {
    BfFrag a, b;
    a.h[0] = *(const ushort8*)(arow + k0 + half * 8);
    a.h[1] = *(const ushort8*)(arow + k0 + 16 + half * 8);
    b.h[0] = *(const ushort8*)(bcol + k0 + half * 8);
    b.h[1] = *(const ushort8*)(bcol + k0 + 16 + half * 8);
    acc = __builtin_amdgcn_wmma_f32_16x16x32_bf16(false, a.v, false, b.v,
                                                  (short)0, acc, false, false);
  }
  const float bv = bias[n0 + l16];
#pragma unroll
  for (int r = 0; r < 8; ++r) {
    int m = m0 + half * 8 + r;
    float v = acc[r] + bv;
    if (doRelu) v = fmaxf(v, 0.f);
    size_t o = (size_t)m * ED + n0 + l16;
    if (outF) outF[o] = v;
    if (outB) outB[o] = f2bf(v);
  }
}

// ---------------- row-wise L2 normalize ----------------
__global__ void l2norm_kernel(const float* __restrict__ fv, float* __restrict__ out) {
  const int n = blockIdx.x, e = threadIdx.x;
  float v = fv[(size_t)n * ED + e];
  float s = v * v;
#pragma unroll
  for (int off = 16; off > 0; off >>= 1) s += __shfl_xor(s, off, 32);
  __shared__ float part[8];
  if ((e & 31) == 0) part[e >> 5] = s;
  __syncthreads();
  float tot = 0.f;
#pragma unroll
  for (int i = 0; i < 8; ++i) tot += part[i];
  out[(size_t)n * ED + e] = v * rsqrtf(tot);
}

extern "C" void kernel_launch(void* const* d_in, const int* in_sizes, int n_in,
                              void* d_out, int out_size, void* d_ws, size_t ws_size,
                              hipStream_t stream) {
  (void)in_sizes; (void)n_in; (void)out_size; (void)ws_size;
  const int*   code   = (const int*)d_in[0];
  const float* aux    = (const float*)d_in[1];
  const int*   length = (const int*)d_in[2];
  // d_in[3] = is_training (inference path only)
  const float* Wlin = (const float*)d_in[4];
  const float* blin = (const float*)d_in[5];
  const float* Kf   = (const float*)d_in[6];
  const float* Rf   = (const float*)d_in[7];
  const float* bl   = (const float*)d_in[8];
  const float* W0   = (const float*)d_in[9];
  const float* b0   = (const float*)d_in[10];
  const float* W1   = (const float*)d_in[11];
  const float* b1   = (const float*)d_in[12];
  float* out = (float*)d_out;

  char* ws = (char*)d_ws; size_t off = 0;
  auto carve = [&](size_t bytes) { char* p = ws + off; off += (bytes + 511) & ~(size_t)511; return p; };
  unsigned short* x_bf  = (unsigned short*)carve((size_t)NB * TT * ED * 2);  // 52.4 MB
  unsigned short* Kt    = (unsigned short*)carve((size_t)G4 * ED * 2);
  unsigned short* Rt    = (unsigned short*)carve((size_t)G4 * ED * 2);
  unsigned short* W0t   = (unsigned short*)carve((size_t)ED * ED * 2);
  unsigned short* W1t   = (unsigned short*)carve((size_t)ED * ED * 2);
  unsigned short* Wb    = (unsigned short*)carve((size_t)536 * ED * 2);
  unsigned short* hbuf0 = (unsigned short*)carve((size_t)NB * ED * 2);
  unsigned short* hbuf1 = (unsigned short*)carve((size_t)NB * ED * 2);
  float*          cbuf  = (float*)carve((size_t)NB * ED * 4);
  unsigned short* xlast = (unsigned short*)carve((size_t)NB * ED * 2);
  unsigned short* fv1   = (unsigned short*)carve((size_t)NB * ED * 2);
  float*          fv    = (float*)carve((size_t)NB * ED * 4);

  prep_weights<<<1024, 256, 0, stream>>>(Kf, Rf, W0, W1, Wlin, Kt, Rt, W0t, W1t, Wb);
  zero_state<<<NB, 256, 0, stream>>>(cbuf, hbuf0);
  embed_kernel<<<NB * TT, 256, 0, stream>>>(code, aux, Wb, blin, x_bf);

  unsigned short* hb[2] = { hbuf0, hbuf1 };
  for (int t = 0; t < TT; ++t) {
    lstm_step_fused<<<dim3(ED / 32, NB / 64), 256, 0, stream>>>(
        x_bf, t, hb[t & 1], hb[(t + 1) & 1], cbuf, Kt, Rt, bl, length, xlast);
  }
  head_gemm<<<dim3(ED / 64, NB / 32), 256, 0, stream>>>(xlast, W0t, b0, nullptr, fv1, 1);
  head_gemm<<<dim3(ED / 64, NB / 32), 256, 0, stream>>>(fv1, W1t, b1, fv, nullptr, 0);
  l2norm_kernel<<<NB, 256, 0, stream>>>(fv, out);
}
